// RelationalGraphLayer_44178033607358
// MI455X (gfx1250) — matrix operand order repaired and checked by reading, not compiled, not run on hardware
//
#include <hip/hip_runtime.h>
#include <hip/hip_bf16.h>

// ---------------------------------------------------------------------------
// Relational GNN layer for MI455X (gfx1250, wave32, WMMA f32_16x16x32_f16)
//
// Pipeline (all on `stream`):
//   1) zero_ws:        clear agg[3][N][128] accumulator + per-type counters
//   2) convert_weights: fp32 weights -> f16, pre-swizzled into WMMA B-fragment
//                       layout (each lane's 16 halves contiguous = 32B loads)
//   3) partition_edges: bucket edge ids by edge_type (atomic counters)
//   4) edge_msg_kernel: per 16-edge tile of one type: 2-layer MLP via WMMA,
//                       relu, scatter-add into agg via global_atomic_add_f32
//   5) node_update_kernel: per 16-node tile: enc=[relu(x)|agg0|agg1|agg2],
//                       2-layer MLP via WMMA per node type present, masked store
// ---------------------------------------------------------------------------

typedef __attribute__((ext_vector_type(16))) _Float16 v16h;
typedef __attribute__((ext_vector_type(8)))  float    v8f;

#define D_DIM   128
#define HID_DIM 128
#define ENC_DIM 512
#define N_ET    3
#define N_NT    2

// fragment counts: We1/We2: 3 types * 4 kchunks * 8 ntiles = 96 each
//                  Wn1: 2 * 16 * 8 = 256 ; Wn2: 2 * 4 * 8 = 64
#define FR_WE1 96
#define FR_WE2 96
#define FR_WN1 256
#define FR_WN2 64
#define FR_TOT (FR_WE1 + FR_WE2 + FR_WN1 + FR_WN2)   // 512

__device__ __forceinline__ v8f wmma16(v16h a, v16h b, v8f c) {
  // (neg_a, A, neg_b, B, c_mod, C, reuse_a, reuse_b)
  return __builtin_amdgcn_wmma_f32_16x16x32_f16(false, a, false, b,
                                                (short)0, c, false, false);
}

// Build an A fragment (16x32 f16, ISA 7.12.2 layout) from row-major LDS.
// lane<16 : row M=lane,    K = k0 + {0,1,2..7, 16..23}
// lane>=16: row M=lane-16, K = k0 + {8..15, 24..31}
__device__ __forceinline__ v16h load_a_frag(const _Float16* sh, int ldk,
                                            int k0, int lane) {
  const _Float16* row = sh + (size_t)(lane & 15) * ldk;
  int kb = k0 + ((lane >> 4) ? 8 : 0);
  v16h a;
#pragma unroll
  for (int v = 0; v < 8; ++v) {
    int kk = kb + (v < 4 ? 2 * v : 16 + 2 * (v - 4));
    a[2 * v]     = row[kk];
    a[2 * v + 1] = row[kk + 1];
  }
  return a;
}

// ---------------------------------------------------------------------------
__global__ void zero_ws_kernel(float* __restrict__ agg, size_t n,
                               int* __restrict__ counts) {
  size_t i = (size_t)blockIdx.x * blockDim.x + threadIdx.x;
  if (i < 16) counts[i] = 0;
  size_t stride = (size_t)gridDim.x * blockDim.x;
  for (; i < n; i += stride) agg[i] = 0.0f;
}

// ---------------------------------------------------------------------------
// Pre-swizzle fp32 weights into f16 WMMA B-fragments.
// Fragment f = (type*KC + kc)*8 + nt ; per fragment: 32 lanes * 16 halves.
// lane<16 : column n = nt*16+lane,    K = kc*32 + {0..7, 16..23}
// lane>=16: column n = nt*16+lane-16, K = kc*32 + {8..15, 24..31}
__global__ void convert_weights_kernel(const float* __restrict__ We1,
                                       const float* __restrict__ We2,
                                       const float* __restrict__ Wn1,
                                       const float* __restrict__ Wn2,
                                       _Float16* __restrict__ we1f,
                                       _Float16* __restrict__ we2f,
                                       _Float16* __restrict__ wn1f,
                                       _Float16* __restrict__ wn2f) {
  int tid = blockIdx.x * blockDim.x + threadIdx.x;
  int g = tid >> 5, lane = tid & 31;
  if (g >= FR_TOT) return;

  const float* srcW;
  _Float16* dstF;
  int KC, fl;
  if (g < FR_WE1) {
    fl = g; KC = 4;
    srcW = We1 + (size_t)(fl / 32) * D_DIM * HID_DIM;  dstF = we1f;
  } else if (g < FR_WE1 + FR_WE2) {
    fl = g - FR_WE1; KC = 4;
    srcW = We2 + (size_t)(fl / 32) * HID_DIM * D_DIM;  dstF = we2f;
  } else if (g < FR_WE1 + FR_WE2 + FR_WN1) {
    fl = g - FR_WE1 - FR_WE2; KC = 16;
    srcW = Wn1 + (size_t)(fl / 128) * ENC_DIM * HID_DIM; dstF = wn1f;
  } else {
    fl = g - FR_WE1 - FR_WE2 - FR_WN1; KC = 4;
    srcW = Wn2 + (size_t)(fl / 32) * HID_DIM * D_DIM;  dstF = wn2f;
  }
  int fpt = KC * 8;
  int r = fl % fpt;
  int kc = r / 8, nt = r % 8;
  int n  = nt * 16 + (lane & 15);
  int kb = kc * 32 + ((lane >> 4) ? 8 : 0);
  _Float16* dst = dstF + (size_t)fl * 512 + (size_t)lane * 16;
#pragma unroll
  for (int h = 0; h < 16; ++h)
    dst[h] = (_Float16)srcW[(size_t)(kb + (h < 8 ? h : h + 8)) * 128 + n];
}

// ---------------------------------------------------------------------------
__global__ void partition_edges_kernel(const int* __restrict__ etype,
                                       int* __restrict__ counts,
                                       int* __restrict__ elist, int E) {
  int e = blockIdx.x * blockDim.x + threadIdx.x;
  if (e >= E) return;
  int t = etype[e];
  int pos = atomicAdd(&counts[t], 1);
  elist[(size_t)t * E + pos] = e;
}

// ---------------------------------------------------------------------------
// One block = 16 edges of a single type, 128 threads (4 wave32).
// Wave w owns N-tiles {2w, 2w+1} of the 16x128 output tile.
__global__ void edge_msg_kernel(const float* __restrict__ nf,
                                const float* __restrict__ be1,
                                const float* __restrict__ be2,
                                const int* __restrict__ src,
                                const int* __restrict__ dst,
                                const int* __restrict__ counts,
                                const int* __restrict__ elist,
                                const _Float16* __restrict__ we1f,
                                const _Float16* __restrict__ we2f,
                                float* __restrict__ agg,
                                int N, int E) {
  int t = blockIdx.y;
  int tile = blockIdx.x;
  int cnt = counts[t];
  if (tile * 16 >= cnt) return;   // uniform exit, no barriers crossed

  __shared__ _Float16 shA[16 * D_DIM];    // staged src features (f16)
  __shared__ _Float16 shH[16 * HID_DIM];  // layer-1 activations (f16)
  __shared__ int s_src[16], s_dst[16], s_valid[16];

  int tid = threadIdx.x;
  int lane = tid & 31, wave = tid >> 5;

  if (tid < 16) {
    int ei = tile * 16 + tid;
    int valid = ei < cnt;
    int e = valid ? elist[(size_t)t * E + ei] : 0;
    s_valid[tid] = valid;
    s_src[tid] = valid ? src[e] : 0;
    s_dst[tid] = valid ? dst[e] : 0;
  }
  __syncthreads();

  // stage A: 16 x 128, fp32 gather -> f16 LDS
  for (int i = tid; i < 16 * D_DIM; i += 128) {
    int m = i >> 7, k = i & 127;
    float v = s_valid[m] ? nf[(size_t)s_src[m] * D_DIM + k] : 0.0f;
    shA[i] = (_Float16)v;
  }
  __syncthreads();

  // ---- layer 1: h1 = relu(A @ We1[t] + be1[t])
#pragma unroll
  for (int q = 0; q < 2; ++q) {
    int nt = wave * 2 + q;
    v8f c = {};
#pragma unroll
    for (int kc = 0; kc < 4; ++kc) {
      v16h a = load_a_frag(shA, D_DIM, kc * 32, lane);
      v16h b = *(const v16h*)(we1f + (size_t)((t * 4 + kc) * 8 + nt) * 512 +
                              (size_t)lane * 16);
      c = wmma16(a, b, c);
    }
    int ncol = nt * 16 + (lane & 15);
    float bias = be1[t * HID_DIM + ncol];
    int mbase = (lane >> 4) * 8;
#pragma unroll
    for (int j = 0; j < 8; ++j) {
      float v = c[j] + bias;
      v = v > 0.0f ? v : 0.0f;
      shH[(mbase + j) * HID_DIM + ncol] = (_Float16)v;
    }
  }
  __syncthreads();

  // ---- layer 2: msg = relu(h1 @ We2[t] + be2[t]) ; scatter-add into agg
#pragma unroll
  for (int q = 0; q < 2; ++q) {
    int nt = wave * 2 + q;
    v8f c = {};
#pragma unroll
    for (int kc = 0; kc < 4; ++kc) {
      v16h a = load_a_frag(shH, HID_DIM, kc * 32, lane);
      v16h b = *(const v16h*)(we2f + (size_t)((t * 4 + kc) * 8 + nt) * 512 +
                              (size_t)lane * 16);
      c = wmma16(a, b, c);
    }
    int ncol = nt * 16 + (lane & 15);
    float bias = be2[t * D_DIM + ncol];
    int mbase = (lane >> 4) * 8;
#pragma unroll
    for (int j = 0; j < 8; ++j) {
      int m = mbase + j;
      if (s_valid[m]) {
        float v = c[j] + bias;
        v = v > 0.0f ? v : 0.0f;
        atomicAdd(&agg[((size_t)t * N + s_dst[m]) * D_DIM + ncol], v);
      }
    }
  }
}

// ---------------------------------------------------------------------------
// One block = 16 nodes, 128 threads. enc = [relu(x) | agg0 | agg1 | agg2].
__global__ void node_update_kernel(const float* __restrict__ nf,
                                   const float* __restrict__ bn1,
                                   const float* __restrict__ bn2,
                                   const int* __restrict__ ntype,
                                   const float* __restrict__ agg,
                                   const _Float16* __restrict__ wn1f,
                                   const _Float16* __restrict__ wn2f,
                                   float* __restrict__ out, int N) {
  int n0 = blockIdx.x * 16;
  __shared__ _Float16 shE[16 * ENC_DIM];   // 16 KB
  __shared__ _Float16 shH[16 * HID_DIM];   // 4 KB
  __shared__ int s_node[16], s_type[16], s_valid[16];

  int tid = threadIdx.x;
  int lane = tid & 31, wave = tid >> 5;

  if (tid < 16) {
    int id = n0 + tid;
    int valid = id < N;
    s_valid[tid] = valid;
    s_node[tid] = valid ? id : 0;
    s_type[tid] = valid ? ntype[id] : -1;
  }
  __syncthreads();

  // stage enc: 16 x 512
  for (int i = tid; i < 16 * ENC_DIM; i += 128) {
    int m = i >> 9, c = i & 511;
    float v = 0.0f;
    if (s_valid[m]) {
      int id = s_node[m];
      if (c < D_DIM) {
        v = nf[(size_t)id * D_DIM + c];
        v = v > 0.0f ? v : 0.0f;
      } else {
        int slot = (c >> 7) - 1;
        v = agg[((size_t)slot * N + id) * D_DIM + (c & 127)];
      }
    }
    shE[i] = (_Float16)v;
  }
  __syncthreads();

  for (int ty = 0; ty < N_NT; ++ty) {
    // uniform skip if no node of this type in the tile
    int any = 0;
#pragma unroll
    for (int m = 0; m < 16; ++m) any |= (s_type[m] == ty);
    if (!any) continue;

    // ---- layer 1: g1 = relu(enc @ Wn1[ty] + bn1[ty])
#pragma unroll
    for (int q = 0; q < 2; ++q) {
      int nt = wave * 2 + q;
      v8f c = {};
#pragma unroll
      for (int kc = 0; kc < 16; ++kc) {
        v16h a = load_a_frag(shE, ENC_DIM, kc * 32, lane);
        v16h b = *(const v16h*)(wn1f + (size_t)((ty * 16 + kc) * 8 + nt) * 512 +
                                (size_t)lane * 16);
        c = wmma16(a, b, c);
      }
      int ncol = nt * 16 + (lane & 15);
      float bias = bn1[ty * HID_DIM + ncol];
      int mbase = (lane >> 4) * 8;
#pragma unroll
      for (int j = 0; j < 8; ++j) {
        float v = c[j] + bias;
        v = v > 0.0f ? v : 0.0f;
        shH[(mbase + j) * HID_DIM + ncol] = (_Float16)v;
      }
    }
    __syncthreads();

    // ---- layer 2: out = g1 @ Wn2[ty] + bn2[ty]  (no relu), masked store
#pragma unroll
    for (int q = 0; q < 2; ++q) {
      int nt = wave * 2 + q;
      v8f c = {};
#pragma unroll
      for (int kc = 0; kc < 4; ++kc) {
        v16h a = load_a_frag(shH, HID_DIM, kc * 32, lane);
        v16h b = *(const v16h*)(wn2f + (size_t)((ty * 4 + kc) * 8 + nt) * 512 +
                                (size_t)lane * 16);
        c = wmma16(a, b, c);
      }
      int ncol = nt * 16 + (lane & 15);
      float bias = bn2[ty * D_DIM + ncol];
      int mbase = (lane >> 4) * 8;
#pragma unroll
      for (int j = 0; j < 8; ++j) {
        int m = mbase + j;
        if (s_valid[m] && s_type[m] == ty)
          out[(size_t)s_node[m] * D_DIM + ncol] = c[j] + bias;
      }
    }
    __syncthreads();   // shH reused by next type's layer-1
  }
}

// ---------------------------------------------------------------------------
extern "C" void kernel_launch(void* const* d_in, const int* in_sizes, int n_in,
                              void* d_out, int out_size, void* d_ws,
                              size_t ws_size, hipStream_t stream) {
  const float* nf  = (const float*)d_in[0];
  const float* We1 = (const float*)d_in[1];
  const float* be1 = (const float*)d_in[2];
  const float* We2 = (const float*)d_in[3];
  const float* be2 = (const float*)d_in[4];
  const float* Wn1 = (const float*)d_in[5];
  const float* bn1 = (const float*)d_in[6];
  const float* Wn2 = (const float*)d_in[7];
  const float* bn2 = (const float*)d_in[8];
  const int* src   = (const int*)d_in[9];
  const int* dst   = (const int*)d_in[10];
  const int* etype = (const int*)d_in[11];
  const int* ntype = (const int*)d_in[12];
  float* out = (float*)d_out;

  int N = in_sizes[0] / D_DIM;
  int E = in_sizes[9];

  // ---- workspace carve-up
  auto align256 = [](size_t x) { return (x + 255) & ~(size_t)255; };
  char* ws = (char*)d_ws;
  size_t off = 0;
  int* counts = (int*)(ws + off);               off = align256(off + 16 * sizeof(int));
  int* elist  = (int*)(ws + off);               off = align256(off + (size_t)N_ET * E * sizeof(int));
  float* agg  = (float*)(ws + off);             off = align256(off + (size_t)N_ET * N * D_DIM * sizeof(float));
  _Float16* we1f = (_Float16*)(ws + off);       off = align256(off + (size_t)FR_WE1 * 512 * sizeof(_Float16));
  _Float16* we2f = (_Float16*)(ws + off);       off = align256(off + (size_t)FR_WE2 * 512 * sizeof(_Float16));
  _Float16* wn1f = (_Float16*)(ws + off);       off = align256(off + (size_t)FR_WN1 * 512 * sizeof(_Float16));
  _Float16* wn2f = (_Float16*)(ws + off);       off = align256(off + (size_t)FR_WN2 * 512 * sizeof(_Float16));
  (void)ws_size; (void)n_in; (void)out_size;

  // 1) zero accumulators + counters
  zero_ws_kernel<<<1024, 256, 0, stream>>>(agg, (size_t)N_ET * N * D_DIM, counts);
  // 2) weight pre-swizzle fp32 -> f16 WMMA fragments
  convert_weights_kernel<<<(FR_TOT * 32 + 255) / 256, 256, 0, stream>>>(
      We1, We2, Wn1, Wn2, we1f, we2f, wn1f, wn2f);
  // 3) bucket edges by type
  partition_edges_kernel<<<(E + 255) / 256, 256, 0, stream>>>(etype, counts, elist, E);
  // 4) edge message MLP + scatter
  dim3 egrid((E + 15) / 16, N_ET);
  edge_msg_kernel<<<egrid, 128, 0, stream>>>(nf, be1, be2, src, dst, counts,
                                             elist, we1f, we2f, agg, N, E);
  // 5) node update MLP
  node_update_kernel<<<(N + 15) / 16, 128, 0, stream>>>(nf, bn1, bn2, ntype,
                                                        agg, wn1f, wn2f, out, N);
}
